// MultiHeadAttention_53369263620708
// MI455X (gfx1250) — matrix operand order repaired and checked by reading, not compiled, not run on hardware
//
#include <hip/hip_runtime.h>
#include <hip/hip_bf16.h>

// MSRSA multi-head attention for MI455X (gfx1250), wave32 + WMMA bf16 path.
// B=8, L=1024, D_IN=256, D_MODEL=512, H=8, D_HEAD=64.

#define B_   8
#define L_   1024
#define DIN  256
#define DM   512
#define H_   8
#define DH   64

typedef __attribute__((ext_vector_type(16))) __bf16 v16bf;
typedef __attribute__((ext_vector_type(8)))  float  v8f;
typedef __attribute__((ext_vector_type(4)))  float  f32x4;

__device__ __forceinline__ __bf16 to_bf16(float f) { return (__bf16)f; }

__device__ __forceinline__ v8f wmma_bf16(v16bf a, v16bf b, v8f c) {
  // D = A(16x32) * B(32x16) + C(16x16 f32)
  return __builtin_amdgcn_wmma_f32_16x16x32_bf16(false, a, false, b, (short)0, c,
                                                 false, false);
}

// ---- fragment loaders -------------------------------------------------------
// A fragment 16x32 bf16 from row-major (ld = leading dim). ISA layout:
// lanes 0-15 row M=lane: K=0..7 then K=16..23; lanes 16-31: K=8..15, 24..31.
__device__ __forceinline__ v16bf load_a_bf16(const __bf16* base, int ld, int lane) {
  const int m = lane & 15, hi = lane >> 4;
  const __bf16* p = base + (size_t)m * ld + hi * 8;
  v16bf a;
  *(f32x4*)&a         = *(const f32x4*)(p);       // K = hi*8 .. +7
  *(((f32x4*)&a) + 1) = *(const f32x4*)(p + 16);  // K = 16+hi*8 .. +7
  return a;
}

// B fragment 32x16 bf16; source stored "N-major": row n holds K contiguous.
// lanes 0-15: N=lane, K=0..7 then 16..23; lanes 16-31: N=lane-16, K=8..15, 24..31.
__device__ __forceinline__ v16bf load_b_bf16(const __bf16* base, int ld, int lane) {
  const int n = lane & 15, hi = lane >> 4;
  const __bf16* p = base + (size_t)n * ld + hi * 8;
  v16bf b;
  *(f32x4*)&b         = *(const f32x4*)(p);
  *(((f32x4*)&b) + 1) = *(const f32x4*)(p + 16);
  return b;
}

__device__ __forceinline__ v16bf load_a_f32(const float* base, int ld, int lane) {
  const int m = lane & 15, hi = lane >> 4;
  const float* p = base + (size_t)m * ld + hi * 8;
  f32x4 x0 = *(const f32x4*)(p);
  f32x4 x1 = *(const f32x4*)(p + 4);
  f32x4 x2 = *(const f32x4*)(p + 16);
  f32x4 x3 = *(const f32x4*)(p + 20);
  v16bf a;
#pragma unroll
  for (int i = 0; i < 4; ++i) {
    a[i] = to_bf16(x0[i]);  a[4 + i] = to_bf16(x1[i]);
    a[8 + i] = to_bf16(x2[i]); a[12 + i] = to_bf16(x3[i]);
  }
  return a;
}

__device__ __forceinline__ v16bf load_b_f32(const float* base, int ld, int lane) {
  const int n = lane & 15, hi = lane >> 4;
  const float* p = base + (size_t)n * ld + hi * 8;
  f32x4 x0 = *(const f32x4*)(p);
  f32x4 x1 = *(const f32x4*)(p + 4);
  f32x4 x2 = *(const f32x4*)(p + 16);
  f32x4 x3 = *(const f32x4*)(p + 20);
  v16bf b;
#pragma unroll
  for (int i = 0; i < 4; ++i) {
    b[i] = to_bf16(x0[i]);  b[4 + i] = to_bf16(x1[i]);
    b[8 + i] = to_bf16(x2[i]); b[12 + i] = to_bf16(x3[i]);
  }
  return b;
}

// ---- kernel 1: fused QKV projection (X @ W^T), fp32 -> bf16 ----------------
// grid: (B*L/128, H, 3); block 256 = 8 waves; wave owns a 16x64 output tile.
// Q,K stored (b,h,l,d) row-major; V stored transposed (b,h,d,l) so that the
// P*V B-fragment later loads contiguously.
__global__ __launch_bounds__(256) void qkv_proj_kernel(
    const float* __restrict__ q_in, const float* __restrict__ k_in,
    const float* __restrict__ v_in, const float* __restrict__ Wq,
    const float* __restrict__ Wk, const float* __restrict__ Wv,
    __bf16* __restrict__ Qws, __bf16* __restrict__ Kws, __bf16* __restrict__ Vtws) {
  const int lane = threadIdx.x & 31;
  const int wave = threadIdx.x >> 5;
  const int which = blockIdx.z;
  const float* X = which == 0 ? q_in : (which == 1 ? k_in : v_in);
  const float* W = which == 0 ? Wq : (which == 1 ? Wk : Wv);
  const int m0 = blockIdx.x * 128 + wave * 16;  // row in flattened (B*L)
  const int h = blockIdx.y;
  const int b = m0 >> 10;
  const int l0 = m0 & (L_ - 1);

  v8f acc[4] = {};
#pragma unroll
  for (int k = 0; k < DIN; k += 32) {
    v16bf a = load_a_f32(X + (size_t)m0 * DIN + k, DIN, lane);
#pragma unroll
    for (int nt = 0; nt < 4; ++nt) {
      v16bf bw = load_b_f32(W + (size_t)(h * DH + nt * 16) * DIN + k, DIN, lane);
      acc[nt] = wmma_bf16(a, bw, acc[nt]);
    }
  }

  const int hi = lane >> 4, n = lane & 15;
  const size_t bh = (size_t)b * H_ + h;
  if (which < 2) {
    __bf16* dst = which == 0 ? Qws : Kws;
#pragma unroll
    for (int nt = 0; nt < 4; ++nt)
#pragma unroll
      for (int g = 0; g < 8; ++g)
        dst[(bh * L_ + (l0 + g + 8 * hi)) * DH + nt * 16 + n] = to_bf16(acc[nt][g]);
  } else {
    // V transposed: contiguous over g -> one 16B store per (nt, lane)
#pragma unroll
    for (int nt = 0; nt < 4; ++nt) {
      __bf16 tmp[8];
#pragma unroll
      for (int g = 0; g < 8; ++g) tmp[g] = to_bf16(acc[nt][g]);
      const int d = nt * 16 + n;
      __bf16* dst = Vtws + (bh * DH + d) * L_ + l0 + 8 * hi;
      *(f32x4*)dst = *(const f32x4*)tmp;
    }
  }
}

// ---- kernel 2: fused MSRSA flash attention ---------------------------------
// grid: (L/16, B); block 256 = 8 waves, wave h == head h.
// Each wave: 16 query rows x full head. adj/dist 16x64 tile staged in LDS once
// per k-step and shared by all 8 heads (8x HBM traffic reduction on the 64 MB
// bias matrices -> the dominant optimization for this workload).
__global__ __launch_bounds__(256) void msrsa_attn_kernel(
    const __bf16* __restrict__ Qws, const __bf16* __restrict__ Kws,
    const __bf16* __restrict__ Vtws, const float* __restrict__ adj,
    const float* __restrict__ dist, const int* __restrict__ amask,
    const float* __restrict__ lambda_a, const float* __restrict__ lambda_d,
    __bf16* __restrict__ AttnOut) {
  __shared__ float s_adj[16 * 64];
  __shared__ float s_dist[16 * 64];
  __shared__ __bf16 s_p[H_][16 * 64];  // per-wave P staging (C-layout -> A-layout)

  const int tid = threadIdx.x;
  const int lane = tid & 31;
  const int h = tid >> 5;
  const int q0 = blockIdx.x * 16;
  const int b = blockIdx.y;
  const size_t bh = (size_t)b * H_ + h;
  const float la = lambda_a[h];
  const float ldm = lambda_d[h];
  const int hi = lane >> 4, n = lane & 15;

  // Preload Q tile as two A fragments (d = 0..31, 32..63)
  v16bf a_q[2];
#pragma unroll
  for (int ks = 0; ks < 2; ++ks)
    a_q[ks] = load_a_bf16(Qws + (bh * L_ + q0) * DH + ks * 32, DH, lane);

  v8f o[4] = {};
  float mrow[8], lrow[8];
#pragma unroll
  for (int g = 0; g < 8; ++g) { mrow[g] = -1e30f; lrow[g] = 0.f; }

  for (int kt = 0; kt < L_ / 64; ++kt) {
    __syncthreads();  // previous iteration's s_adj/s_dist reads done
    {
      const int r = tid >> 4;        // 0..15
      const int c = (tid & 15) * 4;  // 0..60
      const size_t g0 = ((size_t)b * L_ + q0 + r) * L_ + kt * 64 + c;
      *(f32x4*)&s_adj[r * 64 + c] = *(const f32x4*)&adj[g0];
      *(f32x4*)&s_dist[r * 64 + c] = *(const f32x4*)&dist[g0];
    }
    __syncthreads();

    // S = Q K^T (16 x 64 block): 8 WMMAs
    v8f s[4] = {};
#pragma unroll
    for (int nt = 0; nt < 4; ++nt)
#pragma unroll
      for (int ks = 0; ks < 2; ++ks) {
        v16bf bk = load_b_bf16(Kws + (bh * L_ + kt * 64 + nt * 16) * DH + ks * 32,
                               DH, lane);
        s[nt] = wmma_bf16(a_q[ks], bk, s[nt]);
      }

    // scale + graph bias + key mask; track running row max
    float pv[4][8], newmax[8];
#pragma unroll
    for (int g = 0; g < 8; ++g) newmax[g] = mrow[g];
#pragma unroll
    for (int nt = 0; nt < 4; ++nt) {
      const int kk = kt * 64 + nt * 16 + n;
      const float maskadd = (amask[b * L_ + kk] > 0) ? 0.f : -1e9f;
#pragma unroll
      for (int g = 0; g < 8; ++g) {
        const int row = g + 8 * hi;
        float v = s[nt][g] * 0.125f  // 1/sqrt(64)
                  + la * s_adj[row * 64 + nt * 16 + n]
                  + ldm * s_dist[row * 64 + nt * 16 + n] + maskadd;
        pv[nt][g] = v;
        newmax[g] = fmaxf(newmax[g], v);
      }
    }
    // row lives in one 16-lane half -> xor-shuffle reduce (masks 1,2,4,8)
#pragma unroll
    for (int g = 0; g < 8; ++g) {
      float v = newmax[g];
      v = fmaxf(v, __shfl_xor(v, 1, 32));
      v = fmaxf(v, __shfl_xor(v, 2, 32));
      v = fmaxf(v, __shfl_xor(v, 4, 32));
      v = fmaxf(v, __shfl_xor(v, 8, 32));
      newmax[g] = v;
    }
    float rsum[8];
#pragma unroll
    for (int g = 0; g < 8; ++g) rsum[g] = 0.f;
#pragma unroll
    for (int nt = 0; nt < 4; ++nt)
#pragma unroll
      for (int g = 0; g < 8; ++g) {
        float p = __expf(pv[nt][g] - newmax[g]);
        pv[nt][g] = p;
        rsum[g] += p;
      }
#pragma unroll
    for (int g = 0; g < 8; ++g) {
      float v = rsum[g];
      v += __shfl_xor(v, 1, 32);
      v += __shfl_xor(v, 2, 32);
      v += __shfl_xor(v, 4, 32);
      v += __shfl_xor(v, 8, 32);
      const float corr = __expf(mrow[g] - newmax[g]);
      lrow[g] = lrow[g] * corr + v;
      mrow[g] = newmax[g];
      rsum[g] = corr;  // reuse as accumulator rescale
    }
#pragma unroll
    for (int nt = 0; nt < 4; ++nt)
#pragma unroll
      for (int g = 0; g < 8; ++g) o[nt][g] *= rsum[g];

    // P: C-layout f32 -> LDS bf16 -> A-layout fragments (per-wave buffer)
#pragma unroll
    for (int nt = 0; nt < 4; ++nt)
#pragma unroll
      for (int g = 0; g < 8; ++g)
        s_p[h][(g + 8 * hi) * 64 + nt * 16 + n] = to_bf16(pv[nt][g]);

    // O += P * V : 8 WMMAs (V^T stored d-major -> contiguous B fragments)
#pragma unroll
    for (int ks = 0; ks < 2; ++ks) {
      v16bf ap = load_a_bf16(&s_p[h][0] + ks * 32, 64, lane);
#pragma unroll
      for (int nt = 0; nt < 4; ++nt) {
        v16bf bv = load_b_bf16(
            Vtws + (bh * DH + nt * 16) * L_ + kt * 64 + ks * 32, L_, lane);
        o[nt] = wmma_bf16(ap, bv, o[nt]);
      }
    }
  }

  // finalize softmax denominator, store (b, l, h*64+d) as bf16 for out-proj
#pragma unroll
  for (int g = 0; g < 8; ++g) lrow[g] = 1.f / lrow[g];
#pragma unroll
  for (int nt = 0; nt < 4; ++nt)
#pragma unroll
    for (int g = 0; g < 8; ++g) {
      const int row = q0 + g + 8 * hi;
      AttnOut[((size_t)b * L_ + row) * DM + h * DH + nt * 16 + n] =
          to_bf16(o[nt][g] * lrow[g]);
    }
}

// ---- kernel 3: output projection (AttnOut @ W_o^T), bf16 x bf16 -> f32 -----
__global__ __launch_bounds__(256) void out_proj_kernel(
    const __bf16* __restrict__ AttnOut, const float* __restrict__ Wo,
    float* __restrict__ out) {
  const int lane = threadIdx.x & 31;
  const int wave = threadIdx.x >> 5;
  const int m0 = blockIdx.x * 128 + wave * 16;
  const int n0 = blockIdx.y * 64;

  v8f acc[4] = {};
#pragma unroll
  for (int k = 0; k < DM; k += 32) {
    v16bf a = load_a_bf16(AttnOut + (size_t)m0 * DM + k, DM, lane);
#pragma unroll
    for (int nt = 0; nt < 4; ++nt) {
      v16bf bw = load_b_f32(Wo + (size_t)(n0 + nt * 16) * DM + k, DM, lane);
      acc[nt] = wmma_bf16(a, bw, acc[nt]);
    }
  }
  const int hi = lane >> 4, n = lane & 15;
#pragma unroll
  for (int nt = 0; nt < 4; ++nt)
#pragma unroll
    for (int g = 0; g < 8; ++g)
      out[(size_t)(m0 + g + 8 * hi) * DM + n0 + nt * 16 + n] = acc[nt][g];
}

// ---- launcher ---------------------------------------------------------------
extern "C" void kernel_launch(void* const* d_in, const int* in_sizes, int n_in,
                              void* d_out, int out_size, void* d_ws,
                              size_t ws_size, hipStream_t stream) {
  const float* queries = (const float*)d_in[0];
  const float* keys = (const float*)d_in[1];
  const float* values = (const float*)d_in[2];
  const int* amask = (const int*)d_in[3];
  const float* adj = (const float*)d_in[4];
  const float* dist = (const float*)d_in[5];
  const float* Wq = (const float*)d_in[6];
  const float* Wk = (const float*)d_in[7];
  const float* Wv = (const float*)d_in[8];
  const float* Wo = (const float*)d_in[9];
  const float* la = (const float*)d_in[10];
  const float* ld = (const float*)d_in[11];
  float* out = (float*)d_out;

  const size_t perBuf = (size_t)B_ * H_ * L_ * DH;  // 4M bf16 elements (8 MB)
  __bf16* Qws = (__bf16*)d_ws;
  __bf16* Kws = Qws + perBuf;
  __bf16* Vtws = Kws + perBuf;
  __bf16* Attn = Vtws + perBuf;  // (B, L, DM) bf16

  dim3 blk(256);
  qkv_proj_kernel<<<dim3((B_ * L_) / 128, H_, 3), blk, 0, stream>>>(
      queries, keys, values, Wq, Wk, Wv, Qws, Kws, Vtws);
  msrsa_attn_kernel<<<dim3(L_ / 16, B_), blk, 0, stream>>>(
      Qws, Kws, Vtws, adj, dist, amask, la, ld, Attn);
  out_proj_kernel<<<dim3((B_ * L_) / 128, DM / 64), blk, 0, stream>>>(Attn, Wo,
                                                                      out);
}
